// ViTBlock_21277267984716
// MI455X (gfx1250) — compile-verified, matrix operand order checked by reading, not run on hardware
//
#include <hip/hip_runtime.h>
#include <cstdint>

#define EMBED 384
#define HEADS 6
#define HEADD 64
#define FFN 1536
#define BATCH 8
#define SEQ 2048
#define MROWS (BATCH * SEQ)

typedef __attribute__((ext_vector_type(16))) __bf16 v16bf;
typedef __attribute__((ext_vector_type(8))) float v8f;

union FragBF { v16bf v; unsigned u[8]; };

__device__ __forceinline__ unsigned short f2bf(float x) {
  unsigned u = __float_as_uint(x);
  u += 0x7FFFu + ((u >> 16) & 1u);  // round-to-nearest-even
  return (unsigned short)(u >> 16);
}

// Async DMA: global -> LDS, 16B per lane, tracked by ASYNCcnt (CDNA5 §15.18.3).
__device__ __forceinline__ void async_b128(unsigned lds_addr, const void* gaddr) {
  asm volatile("global_load_async_to_lds_b128 %0, %1, off"
               :: "v"(lds_addr), "v"(gaddr) : "memory");
}
__device__ __forceinline__ void async_b128_off16(unsigned lds_addr, const void* gaddr) {
  asm volatile("global_load_async_to_lds_b128 %0, %1, off offset:16"
               :: "v"(lds_addr), "v"(gaddr) : "memory");
}
__device__ __forceinline__ void wait_async0() {
  asm volatile("s_wait_asynccnt 0x0" ::: "memory");
}
__device__ __forceinline__ unsigned lds_addr_of(const void* p) {
  return (unsigned)(uintptr_t)p;  // LDS generic addr: low 32 bits = LDS offset
}

// WMMA fragment gather (ISA 7.12.2): element i -> k = (i%8)+kbase+(i/8)*16,
// kbase = (lane>>4)*8; pairs are contiguous -> compiler merges into ds_load_b128.
__device__ __forceinline__ v16bf load_frag(const unsigned short* row, int kbase) {
  FragBF f;
#pragma unroll
  for (int p = 0; p < 8; ++p) {
    int koff = (p < 4) ? (kbase + 2 * p) : (16 + kbase + 2 * (p - 4));
    f.u[p] = *(const unsigned*)(row + koff);
  }
  return f.v;
}

// ------------------------------------------------ weight convert + transpose
__global__ void convert_transpose_kernel(const float* __restrict__ src,
                                         unsigned short* __restrict__ dst,
                                         int K, int N) {
  int k = blockIdx.x * 32 + (threadIdx.x & 31);
  int n = blockIdx.y * 8 + (threadIdx.x >> 5);
  if (k < K && n < N) dst[(size_t)n * K + k] = f2bf(src[(size_t)k * N + n]);
}

// ---------------------------------------------------------------- rmsnorm
__global__ __launch_bounds__(EMBED) void rmsnorm_kernel(
    const float* __restrict__ x, const float* __restrict__ w,
    unsigned short* __restrict__ out) {
  __shared__ float part[EMBED / 32];
  __shared__ float stot;
  const int row = blockIdx.x;
  const int c = threadIdx.x;
  float v = x[(size_t)row * EMBED + c];
  float p = v * v;
#pragma unroll
  for (int o = 16; o > 0; o >>= 1) p += __shfl_xor(p, o, 32);
  if ((threadIdx.x & 31) == 0) part[threadIdx.x >> 5] = p;
  __syncthreads();
  if (threadIdx.x == 0) {
    float s = 0.f;
#pragma unroll
    for (int i = 0; i < EMBED / 32; ++i) s += part[i];
    stot = s;
  }
  __syncthreads();
  float rs = rsqrtf(stot * (1.0f / EMBED) + 1e-6f);
  out[(size_t)row * EMBED + c] = f2bf(v * rs * w[c]);
}

// ---------------------------------------------------------------- WMMA GEMM
// C[M,N] = act(A[M,K] @ WT[N,K]^T + bias) + resid. Block tile 128x128, 8 waves
// (4x2), each wave 32x64 -> 2 A-frags x 4 B-frags = 8 WMMAs per 32-K step.
// Double-buffered LDS filled with GLOBAL_LOAD_ASYNC_TO_LDS_B128.
// OUTMODE: 0 = f32, 1 = bf16, 2 = bf16 transposed per batch ([B,C,T], for V).
#define AST 40  // LDS row stride (elems): 80B, 16B-aligned, 20-bank skew
template <bool BIAS, bool GELU, bool RESID, int OUTMODE>
__global__ __launch_bounds__(256) void gemm_bf16_kernel(
    const unsigned short* __restrict__ A, const unsigned short* __restrict__ WT,
    const float* __restrict__ bias, const float* __restrict__ resid,
    void* __restrict__ outp, int M, int N, int K) {
  __shared__ unsigned short As[2][128 * AST];
  __shared__ unsigned short Bs[2][128 * AST];
  const int tid = threadIdx.x;
  const int lane = tid & 31;
  const int wave = tid >> 5;
  const int wm = wave >> 1, wn = wave & 1;  // 4x2 waves of 32x64
  const int m0 = blockIdx.y * 128;
  const int n0 = blockIdx.x * 128;
  const int mloc = lane & 15;
  const int kbase = (lane >> 4) * 8;

  auto issue = [&](int k0, int buf) {
    int r = tid >> 1, c = (tid & 1) * 16;
    {  // A tile 128x32
      const unsigned short* g = A + (size_t)(m0 + r) * K + k0 + c;
      unsigned l = lds_addr_of(&As[buf][r * AST + c]);
      async_b128(l, g);
      async_b128_off16(l, g);
    }
    {  // B tile 128x32 from WT[N][K]
      const unsigned short* g = WT + (size_t)(n0 + r) * K + k0 + c;
      unsigned l = lds_addr_of(&Bs[buf][r * AST + c]);
      async_b128(l, g);
      async_b128_off16(l, g);
    }
  };

  v8f acc[2][4] = {};
  const int nk = K >> 5;
  issue(0, 0);
  wait_async0();
  __syncthreads();
  for (int i = 0; i < nk; ++i) {
    const int p = i & 1;
    if (i + 1 < nk) issue((i + 1) << 5, p ^ 1);
    const unsigned short* a0 = &As[p][(wm * 32 + mloc) * AST];
    const unsigned short* b0 = &Bs[p][(wn * 64 + mloc) * AST];
    v16bf fa[2], fb[4];
#pragma unroll
    for (int t = 0; t < 2; ++t) fa[t] = load_frag(a0 + t * 16 * AST, kbase);
#pragma unroll
    for (int u = 0; u < 4; ++u) fb[u] = load_frag(b0 + u * 16 * AST, kbase);
#pragma unroll
    for (int t = 0; t < 2; ++t)
#pragma unroll
      for (int u = 0; u < 4; ++u)
        acc[t][u] = __builtin_amdgcn_wmma_f32_16x16x32_bf16(
            false, fa[t], false, fb[u], (short)0, acc[t][u], false, false);
    if (i + 1 < nk) {
      wait_async0();
      __syncthreads();
    }
  }

#pragma unroll
  for (int t = 0; t < 2; ++t) {
#pragma unroll
    for (int u = 0; u < 4; ++u) {
      const int ncol = n0 + wn * 64 + u * 16 + (lane & 15);
      float bv = BIAS ? bias[ncol] : 0.f;
#pragma unroll
      for (int j = 0; j < 8; ++j) {
        int mrow = m0 + wm * 32 + t * 16 + (lane >> 4) * 8 + j;
        float v = acc[t][u][j];
        if (BIAS) v += bv;
        if (GELU) v = 0.5f * v * (1.0f + erff(v * 0.70710678118654752f));
        if (RESID) v += resid[(size_t)mrow * N + ncol];
        if (OUTMODE == 0) {
          ((float*)outp)[(size_t)mrow * N + ncol] = v;
        } else if (OUTMODE == 1) {
          ((unsigned short*)outp)[(size_t)mrow * N + ncol] = f2bf(v);
        } else {  // V^T layout: [B][C][T]
          int b = mrow >> 11, tt = mrow & (SEQ - 1);
          ((unsigned short*)outp)[((size_t)(b * EMBED + ncol)) * SEQ + tt] = f2bf(v);
        }
      }
    }
  }
}

// ---------------------------------------------------------------- flash attn
// 64 query rows per block, 64-key chunks, double-buffered async K/VT staging,
// online softmax (4 lanes per row), S=QK^T and O+=P@V via WMMA (8/chunk/wave).
#define QST 72   // Qs/Ks/VTs stride: 144B rows, 16B-aligned
__global__ __launch_bounds__(256) void flash_attn_kernel(
    const unsigned short* __restrict__ Q, const unsigned short* __restrict__ Kb,
    const unsigned short* __restrict__ VT, unsigned short* __restrict__ O) {
  __shared__ unsigned short Qs[64 * QST];
  __shared__ unsigned short Ks[2][64 * QST];
  __shared__ unsigned short VTs[2][64 * QST];
  __shared__ unsigned short Ps[64 * 66];
  __shared__ float Ss[64 * 65];
  __shared__ float mrow[64], lrow[64], arow[64];

  const int tid = threadIdx.x, lane = tid & 31, wave = tid >> 5;
  const int wm = wave >> 1, wn = wave & 1;
  const int b = blockIdx.z, h = blockIdx.y;
  const int q0 = blockIdx.x * 64;
  const size_t baseQ = ((size_t)(b * SEQ + q0)) * EMBED + h * HEADD;
  const size_t baseK = ((size_t)(b * SEQ)) * EMBED + h * HEADD;
  const size_t baseVT = ((size_t)(b * EMBED + h * HEADD)) * SEQ;
  const int mloc = lane & 15;
  const int kbase = (lane >> 4) * 8;
  const int rb = wm * 16 + (lane >> 4) * 8;

  auto issueKV = [&](int kc, int buf) {
    int r = tid >> 2, c = (tid & 3) * 16;
    {  // K chunk 64x64 row-major [key][d]
      const unsigned short* g = Kb + baseK + (size_t)(kc + r) * EMBED + c;
      unsigned l = lds_addr_of(&Ks[buf][r * QST + c]);
      async_b128(l, g);
      async_b128_off16(l, g);
    }
    {  // V^T chunk 64(d) x 64(key): straight copy from [B,C,T]
      const unsigned short* g = VT + baseVT + (size_t)r * SEQ + kc + c;
      unsigned l = lds_addr_of(&VTs[buf][r * QST + c]);
      async_b128(l, g);
      async_b128_off16(l, g);
    }
  };

  {  // Q tile 64x64 async
    int r = tid >> 2, c = (tid & 3) * 16;
    const unsigned short* g = Q + baseQ + (size_t)r * EMBED + c;
    unsigned l = lds_addr_of(&Qs[r * QST + c]);
    async_b128(l, g);
    async_b128_off16(l, g);
  }
  issueKV(0, 0);
  if (tid < 64) { mrow[tid] = -1e30f; lrow[tid] = 0.f; }
  wait_async0();
  __syncthreads();

  v8f o0 = {}, o1 = {};
  const int nchunks = SEQ / 64;
  for (int j = 0; j < nchunks; ++j) {
    const int p = j & 1;
    if (j + 1 < nchunks) issueKV((j + 1) * 64, p ^ 1);
    {  // S = Q K^T * 1/sqrt(64): wave tile 16x32 (two 16x16 key tiles)
      v8f s0 = {}, s1 = {};
      const unsigned short* qrow = &Qs[(wm * 16 + mloc) * QST];
      const unsigned short* k0r = &Ks[p][(wn * 32 + mloc) * QST];
      const unsigned short* k1r = k0r + 16 * QST;
#pragma unroll
      for (int kk = 0; kk < 64; kk += 32) {
        v16bf a = load_frag(qrow + kk, kbase);
        s0 = __builtin_amdgcn_wmma_f32_16x16x32_bf16(false, a, false, load_frag(k0r + kk, kbase), (short)0, s0, false, false);
        s1 = __builtin_amdgcn_wmma_f32_16x16x32_bf16(false, a, false, load_frag(k1r + kk, kbase), (short)0, s1, false, false);
      }
      const int col = wn * 32 + (lane & 15);
#pragma unroll
      for (int jj = 0; jj < 8; ++jj) {
        Ss[(rb + jj) * 65 + col] = s0[jj] * 0.125f;
        Ss[(rb + jj) * 65 + col + 16] = s1[jj] * 0.125f;
      }
    }
    __syncthreads();
    {  // online softmax: 4 lanes per query row, 16 cols each
      const int r = tid >> 2, qd = tid & 3;
      const float* srow = &Ss[r * 65 + qd * 16];
      float mo = mrow[r];
      float mx = -1e30f;
#pragma unroll
      for (int c2 = 0; c2 < 16; ++c2) mx = fmaxf(mx, srow[c2]);
      mx = fmaxf(mx, __shfl_xor(mx, 1, 32));
      mx = fmaxf(mx, __shfl_xor(mx, 2, 32));
      float mn = fmaxf(mo, mx);
      float sum = 0.f;
#pragma unroll
      for (int c2 = 0; c2 < 16; ++c2) {
        float pv = __expf(srow[c2] - mn);
        sum += pv;
        Ps[r * 66 + qd * 16 + c2] = f2bf(pv);
      }
      sum += __shfl_xor(sum, 1, 32);
      sum += __shfl_xor(sum, 2, 32);
      if (qd == 0) {
        float al = __expf(mo - mn);
        lrow[r] = lrow[r] * al + sum;
        mrow[r] = mn;
        arow[r] = al;
      }
    }
    __syncthreads();
    {  // rescale accumulators, O += P @ V  (4 WMMAs: 2 key-halves x 2 d-tiles)
#pragma unroll
      for (int jj = 0; jj < 8; ++jj) {
        float al = arow[rb + jj];
        o0[jj] *= al; o1[jj] *= al;
      }
      const unsigned short* prow = &Ps[(wm * 16 + mloc) * 66];
      v16bf pa0 = load_frag(prow, kbase);       // keys 0..31
      v16bf pa1 = load_frag(prow + 32, kbase);  // keys 32..63
      const int nh = wn * 32;
      const unsigned short* v0r = &VTs[p][(nh + mloc) * QST];
      const unsigned short* v1r = v0r + 16 * QST;
      o0 = __builtin_amdgcn_wmma_f32_16x16x32_bf16(false, pa0, false, load_frag(v0r, kbase), (short)0, o0, false, false);
      o0 = __builtin_amdgcn_wmma_f32_16x16x32_bf16(false, pa1, false, load_frag(v0r + 32, kbase), (short)0, o0, false, false);
      o1 = __builtin_amdgcn_wmma_f32_16x16x32_bf16(false, pa0, false, load_frag(v1r, kbase), (short)0, o1, false, false);
      o1 = __builtin_amdgcn_wmma_f32_16x16x32_bf16(false, pa1, false, load_frag(v1r + 32, kbase), (short)0, o1, false, false);
    }
    wait_async0();
    __syncthreads();
  }
  {  // normalize, store bf16 [B,T,C]
    const int nh = wn * 32;
#pragma unroll
    for (int jj = 0; jj < 8; ++jj) {
      int r = rb + jj;
      float inv = 1.0f / lrow[r];
      size_t dst = ((size_t)(b * SEQ + q0 + r)) * EMBED + h * HEADD;
      O[dst + nh + (lane & 15)] = f2bf(o0[jj] * inv);
      O[dst + nh + 16 + (lane & 15)] = f2bf(o1[jj] * inv);
    }
  }
}

// ---------------------------------------------------------------- launch
extern "C" void kernel_launch(void* const* d_in, const int* in_sizes, int n_in,
                              void* d_out, int out_size, void* d_ws,
                              size_t ws_size, hipStream_t stream) {
  const float* x   = (const float*)d_in[0];
  const float* n1w = (const float*)d_in[1];
  const float* n2w = (const float*)d_in[2];
  const float* wq  = (const float*)d_in[3];
  const float* wk  = (const float*)d_in[4];
  const float* wv  = (const float*)d_in[5];
  const float* wo  = (const float*)d_in[6];
  const float* bo  = (const float*)d_in[7];
  const float* w1  = (const float*)d_in[8];
  const float* b1  = (const float*)d_in[9];
  const float* w2  = (const float*)d_in[10];
  const float* b2  = (const float*)d_in[11];
  float* out = (float*)d_out;

  char* ws = (char*)d_ws;
  size_t off = 0;
  auto alloc = [&](size_t bytes) -> void* {
    void* p = ws + off;
    off = (off + bytes + 255) & ~(size_t)255;
    return p;
  };
  const size_t M = MROWS;
  unsigned short* wqT = (unsigned short*)alloc((size_t)EMBED * EMBED * 2);
  unsigned short* wkT = (unsigned short*)alloc((size_t)EMBED * EMBED * 2);
  unsigned short* wvT = (unsigned short*)alloc((size_t)EMBED * EMBED * 2);
  unsigned short* woT = (unsigned short*)alloc((size_t)EMBED * EMBED * 2);
  unsigned short* w1T = (unsigned short*)alloc((size_t)EMBED * FFN * 2);
  unsigned short* w2T = (unsigned short*)alloc((size_t)FFN * EMBED * 2);
  unsigned short* xn  = (unsigned short*)alloc(M * EMBED * 2);
  unsigned short* qb  = (unsigned short*)alloc(M * EMBED * 2);
  unsigned short* kb  = (unsigned short*)alloc(M * EMBED * 2);
  unsigned short* vt  = (unsigned short*)alloc(M * EMBED * 2);  // [B,C,T]
  unsigned short* at  = (unsigned short*)alloc(M * EMBED * 2);
  float*          x1  = (float*)alloc(M * EMBED * 4);
  unsigned short* hb  = (unsigned short*)alloc(M * EMBED * 2);
  unsigned short* f1  = (unsigned short*)alloc(M * FFN * 2);

  auto convT = [&](const float* s, unsigned short* d, int K, int N) {
    convert_transpose_kernel<<<dim3(K / 32, N / 8), 256, 0, stream>>>(s, d, K, N);
  };
  convT(wq, wqT, EMBED, EMBED); convT(wk, wkT, EMBED, EMBED);
  convT(wv, wvT, EMBED, EMBED); convT(wo, woT, EMBED, EMBED);
  convT(w1, w1T, EMBED, FFN);   convT(w2, w2T, FFN, EMBED);

  rmsnorm_kernel<<<M, EMBED, 0, stream>>>(x, n1w, xn);

  dim3 gq(EMBED / 128, M / 128);
  gemm_bf16_kernel<false, false, false, 1><<<gq, 256, 0, stream>>>(
      xn, wqT, nullptr, nullptr, qb, M, EMBED, EMBED);
  gemm_bf16_kernel<false, false, false, 1><<<gq, 256, 0, stream>>>(
      xn, wkT, nullptr, nullptr, kb, M, EMBED, EMBED);
  gemm_bf16_kernel<false, false, false, 2><<<gq, 256, 0, stream>>>(
      xn, wvT, nullptr, nullptr, vt, M, EMBED, EMBED);

  flash_attn_kernel<<<dim3(SEQ / 64, HEADS, BATCH), 256, 0, stream>>>(qb, kb, vt, at);

  gemm_bf16_kernel<true, false, true, 0><<<gq, 256, 0, stream>>>(
      at, woT, bo, x, x1, M, EMBED, EMBED);

  rmsnorm_kernel<<<M, EMBED, 0, stream>>>(x1, n2w, hb);

  gemm_bf16_kernel<true, true, false, 1><<<dim3(FFN / 128, M / 128), 256, 0, stream>>>(
      hb, w1T, b1, nullptr, f1, M, FFN, EMBED);

  gemm_bf16_kernel<true, false, true, 0><<<gq, 256, 0, stream>>>(
      f1, w2T, b2, x1, out, M, EMBED, FFN);
}
